// MLPNetworkWithAttention1__80169859547542
// MI455X (gfx1250) — matrix operand order repaired and checked by reading, not compiled, not run on hardware
//
#include <hip/hip_runtime.h>

typedef __bf16 bf16;
typedef __attribute__((ext_vector_type(16))) __bf16 v16bf;
typedef __attribute__((ext_vector_type(8)))  __bf16 v8bf;
typedef __attribute__((ext_vector_type(8)))  float  v8f;

#define HDEF   256
#define HDIM   32
#define LEAK   0.01f
#define BN_EPS 1e-5f

__device__ __forceinline__ v16bf cat8(v8bf lo, v8bf hi) {
  return __builtin_shufflevector(lo, hi, 0,1,2,3,4,5,6,7,8,9,10,11,12,13,14,15);
}

// WMMA bf16 A fragment (16x32, M=lane%16). Per ISA table:
// lanes 0-15: VGPR0-3 -> K 0..7, VGPR4-7 -> K 16..23
// lanes 16-31: K 8..15 and 24..31. Row data contiguous in K.
__device__ __forceinline__ v16bf a_frag_row(const bf16* row, int kc, int lane) {
  int off = (lane < 16) ? 0 : 8;
  v8bf lo = *(const v8bf*)(row + kc * 32 + off);
  v8bf hi = *(const v8bf*)(row + kc * 32 + off + 16);
  return cat8(lo, hi);
}

// WMMA bf16 B fragment (32x16). lane%16 = column N; lanes 0-15 hold K 0..7,16..23,
// lanes 16-31 hold K 8..15,24..31 (same as A). Weights pre-transposed to
// Wt[n][k] so this is two contiguous 16B loads.
__device__ __forceinline__ v16bf b_frag(const bf16* Wt, int Krows, int nt, int kc, int lane) {
  int col = nt * 16 + (lane & 15);
  int kb  = kc * 32 + ((lane < 16) ? 0 : 8);
  const bf16* r = Wt + (size_t)col * Krows + kb;
  v8bf lo = *(const v8bf*)(r);
  v8bf hi = *(const v8bf*)(r + 16);
  return cat8(lo, hi);
}

__device__ __forceinline__ v8f wmma_bf16(v16bf a, v16bf b, v8f c) {
  return __builtin_amdgcn_wmma_f32_16x16x32_bf16(false, a, false, b, (short)0, c, false, false);
}

// ---------------- stage 0: weight prep ----------------
__global__ void k_transpose_bf16(const float* __restrict__ W, bf16* __restrict__ Wt, int K, int N) {
  int i = blockIdx.x * blockDim.x + threadIdx.x;   // i = n*K + k  (coalesced store)
  if (i >= K * N) return;
  int n = i / K, k = i % K;
  Wt[i] = (bf16)W[(size_t)k * N + n];
}

__global__ void k_zero_stats(float* __restrict__ sum, float* __restrict__ sumsq) {
  int i = blockIdx.x * blockDim.x + threadIdx.x;
  if (i < 512) { sum[i] = 0.f; sumsq[i] = 0.f; }
}

// ---- stage 1: fused embed + Q/K/V + attention + fco, X in bf16 ----
__global__ void __launch_bounds__(32) k_attn(const float* __restrict__ x,
    const float* __restrict__ emb_W, const float* __restrict__ emb_b,
    const bf16* __restrict__ qWt, const bf16* __restrict__ kWt,
    const bf16* __restrict__ vWt, const bf16* __restrict__ fcoWt,
    const float* __restrict__ v_b, const float* __restrict__ fco_b,
    const int* __restrict__ agent_idx, bf16* __restrict__ X) {
  __shared__ float xl[16][40];        // 2.5 KB input tile (padded)
  __shared__ bf16  El[2 * 16 * HDEF]; // 16 KB: e_k for both others; e_q reuses [0:8KB)
  __shared__ bf16  Kl[32 * HDEF];     // 16 KB
  __shared__ bf16  Vl[32 * HDEF];     // 16 KB
  __shared__ bf16  Ql[16 * HDEF];     //  8 KB: Q, then reused for av
  int lane = threadIdx.x;
  int m0 = blockIdx.x * 16;
  int M = lane & 15, hi = lane >> 4;
  int aq = agent_idx[0];
  int ag0 = (aq == 0) ? 1 : 0;
  int ag1 = (aq == 2) ? 1 : 2;

  // ---- load x tile ----
  for (int i = lane; i < 16 * 39; i += 32) {
    int s = i / 39, f = i % 39;
    xl[s][f] = x[(size_t)(m0 + s) * 39 + f];
  }
  __syncthreads();

  // ---- embeddings for the two "other" agents (VALU; K=13 too small for WMMA) ----
  for (int i = lane; i < 2 * 16 * HDEF; i += 32) {   // layout [j][s][h], h fastest
    int h = i & 255, s = (i >> 8) & 15, j = i >> 12;
    int n = j ? ag1 : ag0;
    const float* W = emb_W + (size_t)n * 13 * HDEF;
    float acc = emb_b[n * HDEF + h];
    #pragma unroll
    for (int f = 0; f < 12; ++f) acc = fmaf(xl[s][n * 12 + f], W[f * HDEF + h], acc);
    acc = fmaf(xl[s][36 + n], W[12 * HDEF + h], acc);
    El[i] = (bf16)acc;
  }
  __syncthreads();

  // ---- K and V GEMMs (A from LDS, B streamed from L2) ----
  #pragma unroll
  for (int j = 0; j < 2; ++j) {
    const bf16* rowp = El + (j * 16 + M) * HDEF;
    v16bf A[8];
    #pragma unroll
    for (int kc = 0; kc < 8; ++kc) A[kc] = a_frag_row(rowp, kc, lane);
    for (int nt = 0; nt < 16; ++nt) {
      v8f ck = {}; v8f cv = {};
      #pragma unroll
      for (int kc = 0; kc < 8; ++kc) {
        ck = wmma_bf16(A[kc], b_frag(kWt, HDEF, nt, kc, lane), ck);
        cv = wmma_bf16(A[kc], b_frag(vWt, HDEF, nt, kc, lane), cv);
      }
      int col = nt * 16 + M;
      float vb = v_b[col];
      #pragma unroll
      for (int r = 0; r < 8; ++r) {
        int row = j * 16 + r + 8 * hi;
        Kl[row * HDEF + col] = (bf16)ck[r];
        float v = cv[r] + vb;
        v = (v > 0.f) ? v : LEAK * v;   // leaky_relu on V
        Vl[row * HDEF + col] = (bf16)v;
      }
    }
  }
  __syncthreads();

  // ---- e_q embedding (reuses El[0:8KB)); also the fp->bf16 e_q half of X ----
  for (int i = lane; i < 16 * HDEF; i += 32) {
    int h = i & 255, s = i >> 8;
    const float* W = emb_W + (size_t)aq * 13 * HDEF;
    float acc = emb_b[aq * HDEF + h];
    #pragma unroll
    for (int f = 0; f < 12; ++f) acc = fmaf(xl[s][aq * 12 + f], W[f * HDEF + h], acc);
    acc = fmaf(xl[s][36 + aq], W[12 * HDEF + h], acc);
    bf16 v = (bf16)acc;
    El[i] = v;
    X[(size_t)(m0 + s) * 512 + h] = v;   // e_q half of X
  }
  __syncthreads();

  // ---- Q GEMM ----
  {
    const bf16* rowp = El + M * HDEF;
    v16bf A[8];
    #pragma unroll
    for (int kc = 0; kc < 8; ++kc) A[kc] = a_frag_row(rowp, kc, lane);
    for (int nt = 0; nt < 16; ++nt) {
      v8f cq = {};
      #pragma unroll
      for (int kc = 0; kc < 8; ++kc)
        cq = wmma_bf16(A[kc], b_frag(qWt, HDEF, nt, kc, lane), cq);
      int col = nt * 16 + M;
      #pragma unroll
      for (int r = 0; r < 8; ++r) Ql[(r + 8 * hi) * HDEF + col] = (bf16)cq[r];
    }
  }
  __syncthreads();

  // ---- attention: softmax over 2 keys per (sample, head) ----
  const float scale = 0.17677669529663687f;  // 1/sqrt(32)
  for (int p = 0; p < 4; ++p) {
    int id = p * 32 + lane;        // 16 samples * 8 heads = 128 pairs
    int s = id >> 3, h = id & 7;
    int base = h * HDIM;
    float d0 = 0.f, d1 = 0.f;
    #pragma unroll
    for (int d = 0; d < HDIM; ++d) {
      float q = (float)Ql[s * HDEF + base + d];
      d0 = fmaf(q, (float)Kl[s * HDEF + base + d], d0);
      d1 = fmaf(q, (float)Kl[(16 + s) * HDEF + base + d], d1);
    }
    d0 *= scale; d1 *= scale;
    float m = fmaxf(d0, d1);
    float e0 = __expf(d0 - m), e1 = __expf(d1 - m);
    float inv = 1.f / (e0 + e1);
    float w0 = e0 * inv, w1 = e1 * inv;
    #pragma unroll
    for (int d = 0; d < HDIM; ++d) {
      float av = w0 * (float)Vl[s * HDEF + base + d] + w1 * (float)Vl[(16 + s) * HDEF + base + d];
      Ql[s * HDEF + base + d] = (bf16)av;   // reuse Q slab for av (bf16)
    }
  }
  __syncthreads();

  // ---- fco: attn = av @ fco_W + fco_b, write X[:, 256:512] (bf16) ----
  v16bf A[8];
  #pragma unroll
  for (int kc = 0; kc < 8; ++kc)
    A[kc] = a_frag_row(Ql + M * HDEF, kc, lane);   // av already bf16: no cvt
  for (int nt = 0; nt < 16; ++nt) {
    v8f c = {};
    #pragma unroll
    for (int kc = 0; kc < 8; ++kc)
      c = wmma_bf16(A[kc], b_frag(fcoWt, HDEF, nt, kc, lane), c);
    int col = nt * 16 + M;
    float fb = fco_b[col];
    #pragma unroll
    for (int r = 0; r < 8; ++r) {
      int row = m0 + r + 8 * hi;
      X[(size_t)row * 512 + 256 + col] = (bf16)(c[r] + fb);
    }
  }
}

// ---------------- stage 2: batch-norm statistics ----------------
__global__ void k_stats(const bf16* __restrict__ X, float* __restrict__ sum,
                        float* __restrict__ sumsq) {
  int t = threadIdx.x;               // 256 threads -> features t and t+256
  size_t base = (size_t)blockIdx.x * 128;
  float s0 = 0.f, q0 = 0.f, s1 = 0.f, q1 = 0.f;
  for (int i = 0; i < 128; ++i) {
    const bf16* row = X + (base + i) * 512;
    float a = (float)row[t], b = (float)row[t + 256];
    s0 += a; q0 = fmaf(a, a, q0);
    s1 += b; q1 = fmaf(b, b, q1);
  }
  atomicAdd(&sum[t], s0);        atomicAdd(&sumsq[t], q0);
  atomicAdd(&sum[t + 256], s1);  atomicAdd(&sumsq[t + 256], q1);
}

__global__ void k_finalize(const float* __restrict__ sum, const float* __restrict__ sumsq,
                           float* __restrict__ mrs, int B) {
  int f = blockIdx.x * blockDim.x + threadIdx.x;
  if (f >= 512) return;
  float invB = 1.f / (float)B;
  float mean = sum[f] * invB;
  float var  = sumsq[f] * invB - mean * mean;
  mrs[f]       = mean;
  mrs[512 + f] = rsqrtf(var + BN_EPS);
}

// ---------------- stage 3: BN + fc1 (WMMA) + fc2 ----------------
__global__ void __launch_bounds__(32) k_head(const bf16* __restrict__ X,
    const float* __restrict__ mrs, const bf16* __restrict__ fc1Wt,
    const float* __restrict__ fc1_b, const float* __restrict__ fc2_W,
    const float* __restrict__ fc2_b, float* __restrict__ out) {
  __shared__ float mean_s[512];
  __shared__ float rs_s[512];
  int lane = threadIdx.x;
  for (int i = lane; i < 512; i += 32) { mean_s[i] = mrs[i]; rs_s[i] = mrs[512 + i]; }
  __syncthreads();

  int m0 = blockIdx.x * 16;
  int M = lane & 15, hi = lane >> 4;
  const bf16* row = X + (size_t)(m0 + M) * 512;
  int off = (lane < 16) ? 0 : 8;

  v16bf A[16];                      // normalized Xn, bf16, 16 K-chunks resident
  #pragma unroll
  for (int kc = 0; kc < 16; ++kc) {
    v16bf a;
    #pragma unroll
    for (int i = 0; i < 8; ++i) {
      int f0 = kc * 32 + off + i;
      int f1 = f0 + 16;
      a[i]     = (bf16)(((float)row[f0] - mean_s[f0]) * rs_s[f0]);
      a[8 + i] = (bf16)(((float)row[f1] - mean_s[f1]) * rs_s[f1]);
    }
    A[kc] = a;
  }

  float outacc[8];
  #pragma unroll
  for (int r = 0; r < 8; ++r) outacc[r] = 0.f;

  for (int nt = 0; nt < 16; ++nt) {
    v8f c = {};
    #pragma unroll
    for (int kc = 0; kc < 16; ++kc)
      c = wmma_bf16(A[kc], b_frag(fc1Wt, 512, nt, kc, lane), c);
    int col = nt * 16 + M;
    float b1 = fc1_b[col];
    float w2 = fc2_W[col];
    #pragma unroll
    for (int r = 0; r < 8; ++r) {
      float h = c[r] + b1;
      h = (h > 0.f) ? h : LEAK * h;         // leaky
      outacc[r] = fmaf(h, w2, outacc[r]);   // fc2 partial dot
    }
  }
  float b2 = fc2_b[0];
  #pragma unroll
  for (int r = 0; r < 8; ++r) {
    float v = outacc[r];
    v += __shfl_xor(v, 1);
    v += __shfl_xor(v, 2);
    v += __shfl_xor(v, 4);
    v += __shfl_xor(v, 8);              // reduce over the 16 N-lanes per half
    if (M == 0) out[m0 + r + 8 * hi] = v + b2;
  }
}

extern "C" void kernel_launch(void* const* d_in, const int* in_sizes, int n_in,
                              void* d_out, int out_size, void* d_ws, size_t ws_size,
                              hipStream_t stream) {
  const float* x     = (const float*)d_in[0];
  const float* emb_W = (const float*)d_in[1];
  const float* emb_b = (const float*)d_in[2];
  const float* q_W   = (const float*)d_in[3];
  const float* k_W   = (const float*)d_in[4];
  const float* v_W   = (const float*)d_in[5];
  const float* v_b   = (const float*)d_in[6];
  const float* fco_W = (const float*)d_in[7];
  const float* fco_b = (const float*)d_in[8];
  const float* fc1_W = (const float*)d_in[9];
  const float* fc1_b = (const float*)d_in[10];
  const float* fc2_W = (const float*)d_in[11];
  const float* fc2_b = (const float*)d_in[12];
  const int* agent_idx = (const int*)d_in[13];

  int B = in_sizes[0] / 39;

  char* ws = (char*)d_ws;
  size_t o = 0;
  auto alloc = [&](size_t bytes) -> char* {
    o = (o + 255) & ~(size_t)255;
    char* p = ws + o;
    o += bytes;
    return p;
  };
  bf16*  X     = (bf16*) alloc((size_t)B * 512 * sizeof(bf16));
  bf16*  qWt   = (bf16*) alloc(256 * 256 * sizeof(bf16));
  bf16*  kWt   = (bf16*) alloc(256 * 256 * sizeof(bf16));
  bf16*  vWt   = (bf16*) alloc(256 * 256 * sizeof(bf16));
  bf16*  fcoWt = (bf16*) alloc(256 * 256 * sizeof(bf16));
  bf16*  fc1Wt = (bf16*) alloc(512 * 256 * sizeof(bf16));
  float* sum   = (float*)alloc(512 * sizeof(float));
  float* sumsq = (float*)alloc(512 * sizeof(float));
  float* mrs   = (float*)alloc(1024 * sizeof(float));

  // stage 0: prep
  k_zero_stats<<<2, 256, 0, stream>>>(sum, sumsq);
  k_transpose_bf16<<<(256 * 256 + 255) / 256, 256, 0, stream>>>(q_W,   qWt,   256, 256);
  k_transpose_bf16<<<(256 * 256 + 255) / 256, 256, 0, stream>>>(k_W,   kWt,   256, 256);
  k_transpose_bf16<<<(256 * 256 + 255) / 256, 256, 0, stream>>>(v_W,   vWt,   256, 256);
  k_transpose_bf16<<<(256 * 256 + 255) / 256, 256, 0, stream>>>(fco_W, fcoWt, 256, 256);
  k_transpose_bf16<<<(512 * 256 + 255) / 256, 256, 0, stream>>>(fc1_W, fc1Wt, 512, 256);

  // stage 1: fused embed + Q/K/V + attention + fco -> X (bf16)
  k_attn<<<B / 16, 32, 0, stream>>>(x, emb_W, emb_b, qWt, kWt, vWt, fcoWt,
                                    v_b, fco_b, agent_idx, X);

  // stage 2: batch-norm statistics
  k_stats<<<B / 128, 256, 0, stream>>>(X, sum, sumsq);
  k_finalize<<<2, 256, 0, stream>>>(sum, sumsq, mrs, B);

  // stage 3: BN + fc1 + fc2 -> out
  k_head<<<B / 16, 32, 0, stream>>>(X, mrs, fc1Wt, fc1_b, fc2_W, fc2_b, (float*)d_out);
}
